// gcn_80590766342917
// MI455X (gfx1250) — matrix operand order repaired and checked by reading, not compiled, not run on hardware
//
#include <hip/hip_runtime.h>
#include <hip/hip_bf16.h>

typedef __attribute__((ext_vector_type(16))) __bf16 v16bf;
typedef __attribute__((ext_vector_type(8)))  float  v8f;
typedef __attribute__((ext_vector_type(4)))  float  f4;

#define N_ENT  5000
#define DIM    256
#define BATCH  50
#define NCHUNK 8
#define CHUNK  625   // 5000 / 8

// ---------------------------------------------------------------------------
// Kernel 1: partial column sums  partial[b][c][d] = sum_{n in chunk c} emb[b,n,d]
// ---------------------------------------------------------------------------
__global__ __launch_bounds__(256) void colsum_partial_kernel(
    const float* __restrict__ emb, float* __restrict__ partial) {
  int b = blockIdx.x, c = blockIdx.y, d = threadIdx.x;
  const float* p = emb + ((size_t)b * N_ENT + (size_t)c * CHUNK) * DIM + d;
  float s = 0.f;
  for (int n = 0; n < CHUNK; ++n) s += p[(size_t)n * DIM];
  partial[((size_t)b * NCHUNK + c) * DIM + d] = s;
}

// ---------------------------------------------------------------------------
// Kernel 2: t[b][a] = dot(total[b], W[a]) + bias[a]   (exact f32)
// ---------------------------------------------------------------------------
__global__ __launch_bounds__(256) void compute_tb_kernel(
    const float* __restrict__ partial, const float* __restrict__ W,
    const float* __restrict__ bias, float* __restrict__ tb) {
  __shared__ float tot[DIM];
  int b = blockIdx.x, a = threadIdx.x;
  float s = 0.f;
  for (int c = 0; c < NCHUNK; ++c) s += partial[((size_t)b * NCHUNK + c) * DIM + a];
  tot[a] = s;
  __syncthreads();
  float acc = bias[a];
  const float* wr = W + (size_t)a * DIM;
  for (int d = 0; d < DIM; ++d) acc += tot[d] * wr[d];
  tb[(size_t)b * DIM + a] = acc;
}

// ---------------------------------------------------------------------------
// Kernel 3: pack W (f32 row-major [A,D]) into per-lane bf16 WMMA B-fragments.
// B packing used here (contiguous k, same order as A): lanes 0-15 -> col N=lane, K=kbase+0..15;
// lanes 16-31 -> col N=lane-16, K=kbase+16..31; 2 bf16 per VGPR, K ascending.
// Fragment index: [kk][ct][lane] -> 32 bytes each. Total 8*16*32*32 = 128 KB.
// ---------------------------------------------------------------------------
__global__ __launch_bounds__(32) void pack_w_kernel(
    const float* __restrict__ W, v16bf* __restrict__ bfrag) {
  int ct = blockIdx.x, kk = blockIdx.y, l = threadIdx.x;
  int a    = ct * 16 + (l & 15);
  int koff = kk * 32 + ((l >= 16) ? 16 : 0);
  const float* src = W + (size_t)a * DIM + koff;
  v16bf v;
#pragma unroll
  for (int i = 0; i < 16; ++i) v[i] = (__bf16)src[i];
  bfrag[((size_t)kk * 16 + ct) * 32 + l] = v;
}

// ---------------------------------------------------------------------------
// Kernel 4: main GEMM.  out[b,n,a] = t[b,a] - (emb[b,n] . W[a])
// Workgroup: 512 threads = 16 waves -> 128 rows x 256 cols.
//   wave = rowGroup(0..7) x colHalf(0..1); each wave: 16 rows x 128 cols
//   = 8 accumulator tiles of 16x16 f32 (v8f each).
// All of W (packed fragments, 128 KB) staged in dynamic LDS once per block;
// B fragments double-buffered in registers so ds_load overlaps the WMMA.
// ---------------------------------------------------------------------------
__global__ __launch_bounds__(512) void gemm_wmma_kernel(
    const float* __restrict__ emb, const v16bf* __restrict__ bfrag_g,
    const float* __restrict__ tb, float* __restrict__ out) {
  extern __shared__ char lds_raw[];          // 131072 bytes
  v16bf* ldsB = (v16bf*)lds_raw;

  int tid = threadIdx.x;

  // Cooperative copy of all B fragments: 8192 float4 / 512 threads = 16 each.
  {
    const f4* src = (const f4*)bfrag_g;
    f4*       dst = (f4*)lds_raw;
#pragma unroll
    for (int i = 0; i < 16; ++i) dst[tid + i * 512] = src[tid + i * 512];
  }
  __syncthreads();

  int wave     = tid >> 5;
  int lane     = tid & 31;
  int rowGroup = wave >> 1;
  int colHalf  = wave & 1;
  int b        = blockIdx.y;
  int mBase    = blockIdx.x * 128 + rowGroup * 16;

  int laneRow = lane & 15;
  int laneHi  = lane >> 4;                   // 0: K 0-15, 1: K 16-31
  int r = mBase + laneRow;
  if (r > N_ENT - 1) r = N_ENT - 1;          // clamp OOB reads (stores masked)
  const float* arow = emb + ((size_t)b * N_ENT + r) * DIM;

  // Per-wave base of its 8 column tiles in the fragment array.
  // Index of fragment (kk, t): ((kk*16 + colHalf*8 + t)*32 + lane)
  int fragBase = (colHalf * 8) * 32 + lane;

  v8f acc[8] = {};

#pragma unroll
  for (int kk = 0; kk < 8; ++kk) {
    int baseD = kk * 32 + laneHi * 16;
    f4 a0 = *(const f4*)(arow + baseD);
    f4 a1 = *(const f4*)(arow + baseD + 4);
    f4 a2 = *(const f4*)(arow + baseD + 8);
    f4 a3 = *(const f4*)(arow + baseD + 12);
    v16bf af;
    af[0]  = (__bf16)a0.x;  af[1]  = (__bf16)a0.y;
    af[2]  = (__bf16)a0.z;  af[3]  = (__bf16)a0.w;
    af[4]  = (__bf16)a1.x;  af[5]  = (__bf16)a1.y;
    af[6]  = (__bf16)a1.z;  af[7]  = (__bf16)a1.w;
    af[8]  = (__bf16)a2.x;  af[9]  = (__bf16)a2.y;
    af[10] = (__bf16)a2.z;  af[11] = (__bf16)a2.w;
    af[12] = (__bf16)a3.x;  af[13] = (__bf16)a3.y;
    af[14] = (__bf16)a3.z;  af[15] = (__bf16)a3.w;

    int kBase = kk * 16 * 32 + fragBase;
    // Register double-buffer for B fragments: prefetch t+1 before WMMA(t).
    v16bf bCur = ldsB[kBase];
#pragma unroll
    for (int t = 0; t < 8; ++t) {
      v16bf bNext;
      if (t < 7) bNext = ldsB[kBase + (t + 1) * 32];
      acc[t] = __builtin_amdgcn_wmma_f32_16x16x32_bf16(
          /*neg_a=*/false, af, /*neg_b=*/false, bCur,
          /*c_mod=*/(short)0, acc[t], /*reuse_a=*/false, /*reuse_b=*/false);
      bCur = bNext;
    }
  }

  // Store: C/D layout — lane&15 = column N, VGPR j = row M=j (lanes 0-15) or
  // M=j+8 (lanes 16-31).  out = t[b,a] - acc.
#pragma unroll
  for (int t = 0; t < 8; ++t) {
    int col = (colHalf * 8 + t) * 16 + laneRow;
    float tv = tb[(size_t)b * DIM + col];
#pragma unroll
    for (int j = 0; j < 8; ++j) {
      int row = mBase + j + laneHi * 8;
      if (row < N_ENT)
        out[((size_t)b * N_ENT + row) * DIM + col] = tv - acc[t][j];
    }
  }
}

// ---------------------------------------------------------------------------
extern "C" void kernel_launch(void* const* d_in, const int* in_sizes, int n_in,
                              void* d_out, int out_size, void* d_ws, size_t ws_size,
                              hipStream_t stream) {
  const float* emb  = (const float*)d_in[0];   // [50,5000,256] f32
  const float* W    = (const float*)d_in[1];   // [256,256] f32
  const float* bias = (const float*)d_in[2];   // [256] f32
  float*       out  = (float*)d_out;           // [50,5000,256] f32

  char* ws = (char*)d_ws;
  v16bf* bfrag   = (v16bf*)ws;                       // 131072 B
  float* partial = (float*)(ws + 131072);            // 50*8*256*4 = 409600 B
  float* tb      = (float*)(ws + 131072 + 409600);   // 50*256*4   =  51200 B

  // Allow 128 KB dynamic LDS (gfx1250 WGP has 320 KB). Host-side, capture-safe.
  hipFuncSetAttribute((const void*)gemm_wmma_kernel,
                      hipFuncAttributeMaxDynamicSharedMemorySize, 131072);

  pack_w_kernel<<<dim3(16, 8), 32, 0, stream>>>(W, bfrag);
  colsum_partial_kernel<<<dim3(BATCH, NCHUNK), 256, 0, stream>>>(emb, partial);
  compute_tb_kernel<<<BATCH, 256, 0, stream>>>(partial, W, bias, tb);

  dim3 grid((N_ENT + 127) / 128, BATCH);             // 40 x 50
  gemm_wmma_kernel<<<grid, 512, 131072, stream>>>(emb, bfrag, tb, out);
}
